// InferTree_51599737094739
// MI455X (gfx1250) — matrix operand order repaired and checked by reading, not compiled, not run on hardware
//
#include <hip/hip_runtime.h>
#include <hip/hip_bf16.h>
#include <math.h>

typedef __bf16 bf16_t;
typedef __attribute__((ext_vector_type(16))) __bf16 v16bf;
typedef __attribute__((ext_vector_type(8)))  float  v8f;

#define B_ROWS 8192
#define D_FEAT 512
#define N_COLS 1110
#define NPAD   1152
#define BM 128
#define BN 64
#define BK 32
#define NT (D_FEAT / BK)   // 16 K-tiles
#define LDA 40   // padded bf16 elements per LDS row (80B stride: 16B-aligned b128, conflict-free dword reads)

// workspace layout (bytes)
#define OFF_XP   0u
#define SZ_XP    (B_ROWS * D_FEAT * 2u)        // 8,388,608
#define OFF_WP   (OFF_XP + SZ_XP)
#define SZ_WP    (NPAD * D_FEAT * 2u)          // 1,179,648
#define OFF_BC   (OFF_WP + SZ_WP)
#define SZ_BC    (NPAD * 4u)
#define OFF_PR   (OFF_BC + SZ_BC)
#define SZ_PR    (B_ROWS * NPAD * 4u)          // 37,748,736
#define OFF_NUM  (OFF_PR + SZ_PR)
#define SZ_NUM   (1120u * 4u)
#define OFF_DEN  (OFF_NUM + SZ_NUM)

struct bf4 { bf16_t a, b, c, d; };

// ---- CDNA5 async global->LDS DMA (16B per lane, tracked by ASYNCcnt) ----
__device__ __forceinline__ void async_load_b128(unsigned lds_off, const bf16_t* gaddr) {
  asm volatile("global_load_async_to_lds_b128 %0, %1, off"
               :: "v"(lds_off), "v"(gaddr)
               : "memory");
}

// ---------------- pack x: f32 -> bf16 ----------------
__global__ void pack_x_kernel(const float* __restrict__ x, bf16_t* __restrict__ xp) {
  int i = blockIdx.x * blockDim.x + threadIdx.x;
  if (i >= (B_ROWS * D_FEAT) / 4) return;
  const float4 v = reinterpret_cast<const float4*>(x)[i];
  bf4 o;
  o.a = (bf16_t)v.x; o.b = (bf16_t)v.y; o.c = (bf16_t)v.z; o.d = (bf16_t)v.w;
  reinterpret_cast<bf4*>(xp)[i] = o;
}

// ---------------- pack weights: [n,d,k] f32 -> B-matrix [Npad][512] bf16 ----------------
__global__ void pack_w_kernel(const float* __restrict__ W1, const float* __restrict__ b1,
                              const float* __restrict__ W2, const float* __restrict__ b2,
                              const float* __restrict__ W3, const float* __restrict__ b3,
                              bf16_t* __restrict__ wp, float* __restrict__ bc,
                              float* __restrict__ gnum, float* __restrict__ gden) {
  int idx = blockIdx.x * blockDim.x + threadIdx.x;
  if (idx >= NPAD * D_FEAT) return;
  int n = idx >> 9;        // column (node*10 + k, concatenated over layers)
  int d = idx & 511;       // feature
  float w = 0.f;
  if (n < 10) {
    w = W1[d * 10 + n];                                    // W1[0,d,n]
  } else if (n < 110) {
    int j = n - 10; w = W2[(j / 10) * 5120 + d * 10 + (j % 10)];
  } else if (n < N_COLS) {
    int j = n - 110; w = W3[(j / 10) * 5120 + d * 10 + (j % 10)];
  }
  wp[n * D_FEAT + d] = (bf16_t)w;
  if (d == 0) {
    float bb = 0.f;
    if (n < 10) bb = b1[n];
    else if (n < 110) bb = b2[n - 10];
    else if (n < N_COLS) bb = b3[n - 110];
    bc[n] = bb;
  }
  if (idx < 1120) { gnum[idx] = 0.f; gden[idx] = 0.f; }    // reset accumulators every call
}

// stage one K-tile (A: 128x32, B: 64x32) via async DMA; 3 x b128 per thread
__device__ __forceinline__ void stage_tile_async(
    const bf16_t* __restrict__ xp, const bf16_t* __restrict__ wp,
    int rowBase, int colBase, int k0, unsigned aBase, unsigned bBase, int tid) {
  // A tile: 512 chunks of 16B, 2 per thread
  {
    int c = tid;
    int r = c >> 2, o = (c & 3) * 8;
    async_load_b128(aBase + (unsigned)(r * LDA + o) * 2u,
                    xp + (size_t)(rowBase + r) * D_FEAT + k0 + o);
  }
  {
    int c = tid + 256;
    int r = c >> 2, o = (c & 3) * 8;
    async_load_b128(aBase + (unsigned)(r * LDA + o) * 2u,
                    xp + (size_t)(rowBase + r) * D_FEAT + k0 + o);
  }
  // B tile: 256 chunks of 16B, 1 per thread
  {
    int r = tid >> 2, o = (tid & 3) * 8;
    async_load_b128(bBase + (unsigned)(r * LDA + o) * 2u,
                    wp + (size_t)(colBase + r) * D_FEAT + k0 + o);
  }
}

// ---------------- fused GEMM + bias + sigmoid (bf16 WMMA, async triple-buffered) ----------------
__global__ __launch_bounds__(256) void gemm_sigmoid_kernel(
    const bf16_t* __restrict__ xp, const bf16_t* __restrict__ wp,
    const float* __restrict__ bc, float* __restrict__ probs) {
  __shared__ bf16_t As[3][BM * LDA];
  __shared__ bf16_t Bs[3][BN * LDA];

  const int tid  = threadIdx.x;
  const int lane = tid & 31;
  const int wv   = tid >> 5;        // 8 waves
  const int wm   = wv & 3;          // 4 waves along M (32 rows each)
  const int wn   = wv >> 2;         // 2 waves along N (32 cols each)
  const int m    = lane & 15;
  const int sub  = lane >> 4;       // half-wave selects K-halves
  const int rowBase = blockIdx.x * BM;
  const int colBase = blockIdx.y * BN;

  const unsigned aOff[3] = { (unsigned)(size_t)&As[0][0], (unsigned)(size_t)&As[1][0],
                             (unsigned)(size_t)&As[2][0] };
  const unsigned bOff[3] = { (unsigned)(size_t)&Bs[0][0], (unsigned)(size_t)&Bs[1][0],
                             (unsigned)(size_t)&Bs[2][0] };

  v8f acc[2][2] = {};               // 2x2 WMMA 16x16 tiles per wave

  // prologue: DMA tiles 0 and 1 (prefetch distance 2)
  stage_tile_async(xp, wp, rowBase, colBase, 0 * BK, aOff[0], bOff[0], tid);
  stage_tile_async(xp, wp, rowBase, colBase, 1 * BK, aOff[1], bOff[1], tid);

  for (int kt = 0; kt < NT; ++kt) {
    const int cur = kt % 3;
    // in-flight: tiles kt, kt+1 (3 async ops each, in-order completion)
    if (kt < NT - 1) asm volatile("s_wait_asynccnt 0x3" ::: "memory");
    else             asm volatile("s_wait_asynccnt 0x0" ::: "memory");
    __syncthreads();  // tile kt visible to all waves; all waves done reading buf[(kt+2)%3]

    if (kt + 2 < NT) {
      stage_tile_async(xp, wp, rowBase, colBase, (kt + 2) * BK,
                       aOff[(kt + 2) % 3], bOff[(kt + 2) % 3], tid);
    }

    const bf16_t* Ac = &As[cur][0];
    const bf16_t* Bc = &Bs[cur][0];
    union Frag { unsigned u[8]; v16bf v; } a[2], b[2];
    #pragma unroll
    for (int t = 0; t < 2; ++t) {
      // A fragment: lane holds row m, K-halves per documented 16-bit A 16x32 layout
      int ar = wm * 32 + t * 16 + m;
      #pragma unroll
      for (int j = 0; j < 8; ++j) {
        int g = j >> 2, q = j & 3;
        int k = g * 16 + sub * 8 + 2 * q;
        a[t].u[j] = *reinterpret_cast<const unsigned*>(&Ac[ar * LDA + k]);
      }
      // B fragment: lane holds column m, K-halves matching the A fragment layout
      int br = wn * 32 + t * 16 + m;
      #pragma unroll
      for (int j = 0; j < 8; ++j) {
        int k = (j >> 2) * 16 + sub * 8 + 2 * (j & 3);
        b[t].u[j] = *reinterpret_cast<const unsigned*>(&Bc[br * LDA + k]);
      }
    }
    #pragma unroll
    for (int i = 0; i < 2; ++i)
      #pragma unroll
      for (int j = 0; j < 2; ++j)
        acc[i][j] = __builtin_amdgcn_wmma_f32_16x16x32_bf16(
            false, a[i].v, false, b[j].v, (short)0, acc[i][j], false, false);
    // no trailing barrier: next iteration's start barrier protects buffer reuse
  }

  // epilogue: bias + sigmoid, C/D layout: lane<16 -> M=r, lane>=16 -> M=8+r; N=lane&15
  #pragma unroll
  for (int i = 0; i < 2; ++i) {
    #pragma unroll
    for (int j = 0; j < 2; ++j) {
      int col = colBase + wn * 32 + j * 16 + m;
      float bias = bc[col];
      #pragma unroll
      for (int r = 0; r < 8; ++r) {
        int row = rowBase + wm * 32 + i * 16 + sub * 8 + r;
        float v = acc[i][j][r] + bias;
        probs[(size_t)row * NPAD + col] = 1.0f / (1.0f + __expf(-v));
      }
    }
  }
}

// ---------------- tree products + per-child num/den reduction ----------------
__global__ __launch_bounds__(256) void tree_eval_kernel(
    const float* __restrict__ probs, float* __restrict__ out,
    float* __restrict__ gnum, float* __restrict__ gden) {
  __shared__ float ns[1120];
  __shared__ float ds[1120];
  const int tid = threadIdx.x;
  for (int i = tid; i < 1120; i += 256) { ns[i] = 0.f; ds[i] = 0.f; }
  __syncthreads();

  const int lane = tid & 31;
  const int wv   = tid >> 5;
  for (int rr = 0; rr < 8; ++rr) {
    int row = blockIdx.x * 64 + wv * 8 + rr;
    const float* pr = probs + (size_t)row * NPAD;
    // leaves: out[b,c] = p1 * p2 * p3 (product along path)
    for (int c = lane; c < 1000; c += 32) {
      float p3 = pr[110 + c];
      int   n3 = c / 10;
      float p2 = pr[10 + n3];
      float p1 = pr[n3 / 10];
      float o  = p1 * p2 * p3;
      out[(size_t)row * 1000 + c] = o;
      atomicAdd(&ns[110 + c], o * p3);   // child_pp * prob
      atomicAdd(&ds[110 + c], o);        // child_pp
    }
    // internal layers 1-2
    for (int j = lane; j < 110; j += 32) {
      float p = pr[j];
      float cp = (j < 10) ? p : p * pr[(j - 10) / 10];
      atomicAdd(&ns[j], cp * p);
      atomicAdd(&ds[j], cp);
    }
  }
  __syncthreads();
  for (int i = tid; i < N_COLS; i += 256) {
    atomicAdd(&gnum[i], ns[i]);
    atomicAdd(&gden[i], ds[i]);
  }
}

// ---------------- penalty scan ----------------
__global__ void finalize_kernel(const float* __restrict__ gnum,
                                const float* __restrict__ gden,
                                float* __restrict__ out_pen) {
  __shared__ float s[111];
  int t = threadIdx.x;
  if (t < 111) {
    float acc = 0.f;
    for (int k = 0; k < 10; ++k) {
      int col = t * 10 + k;
      acc += __logf(gnum[col] / gden[col]);
    }
    s[t] = acc;
  }
  __syncthreads();
  if (t == 0) {
    float p = 0.f;
    for (int i = 0; i < 111; ++i) {
      // m_l = 1 + 0.5 * 2^-l / 10 ; node 0 is layer 1, 1..10 layer 2, 11..110 layer 3
      float ml = (i == 0) ? 1.025f : (i < 11 ? 1.0125f : 1.00625f);
      p = (p + s[i]) * ml;
    }
    *out_pen = -p;
  }
}

extern "C" void kernel_launch(void* const* d_in, const int* in_sizes, int n_in,
                              void* d_out, int out_size, void* d_ws, size_t ws_size,
                              hipStream_t stream) {
  const float* x  = (const float*)d_in[0];
  const float* W1 = (const float*)d_in[1];
  const float* b1 = (const float*)d_in[2];
  const float* W2 = (const float*)d_in[3];
  const float* b2 = (const float*)d_in[4];
  const float* W3 = (const float*)d_in[5];
  const float* b3 = (const float*)d_in[6];
  float* out = (float*)d_out;
  char*  ws  = (char*)d_ws;

  bf16_t* xp    = (bf16_t*)(ws + OFF_XP);
  bf16_t* wp    = (bf16_t*)(ws + OFF_WP);
  float*  bc    = (float*)(ws + OFF_BC);
  float*  probs = (float*)(ws + OFF_PR);
  float*  gnum  = (float*)(ws + OFF_NUM);
  float*  gden  = (float*)(ws + OFF_DEN);

  pack_x_kernel<<<(B_ROWS * D_FEAT / 4 + 255) / 256, 256, 0, stream>>>(x, xp);
  pack_w_kernel<<<(NPAD * D_FEAT + 255) / 256, 256, 0, stream>>>(
      W1, b1, W2, b2, W3, b3, wp, bc, gnum, gden);

  dim3 gg(B_ROWS / BM, NPAD / BN);   // 64 x 18
  gemm_sigmoid_kernel<<<gg, 256, 0, stream>>>(xp, wp, bc, probs);

  tree_eval_kernel<<<B_ROWS / 64, 256, 0, stream>>>(probs, out, gnum, gden);

  finalize_kernel<<<1, 128, 0, stream>>>(gnum, gden, out + (size_t)B_ROWS * 1000);
}
